// TransformerDecoderLayer3_87746181857988
// MI455X (gfx1250) — compile-verified, matrix operand order checked by reading, not run on hardware
//
#include <hip/hip_runtime.h>

typedef _Float16 h16;
typedef __attribute__((ext_vector_type(16))) _Float16 v16h;
typedef __attribute__((ext_vector_type(8)))  _Float16 v8h;
typedef __attribute__((ext_vector_type(8)))  float    v8f;
typedef int v4i __attribute__((vector_size(16)));

#define AS1 __attribute__((address_space(1)))
#define AS3 __attribute__((address_space(3)))

#define B_   4
#define L_   2048
#define NT_  (B_ * L_)        // 8192 total rows
#define D_   512
#define H_   8
#define DH_  64
#define HID_ 1024
#define CLAMP_ 50000.0f

#if __has_builtin(__builtin_amdgcn_global_load_async_to_lds_b128)
#define USE_ASYNC_LDS 1
#else
#define USE_ASYNC_LDS 0
#endif

// -------------------------------------------------------------------------
// helpers
// -------------------------------------------------------------------------
__device__ __forceinline__ v8f zero8() {
    v8f z;
#pragma unroll
    for (int i = 0; i < 8; ++i) z[i] = 0.0f;
    return z;
}

__device__ __forceinline__ v8f wmma32(v16h a, v16h b, v8f c) {
    // v_wmma_f32_16x16x32_f16: (neg_a, A, neg_b, B, c_mod, C, reuse_a, reuse_b)
    return __builtin_amdgcn_wmma_f32_16x16x32_f16(false, a, false, b,
                                                  (short)0, c, false, false);
}

// 16-byte global -> LDS copy; async (ASYNCcnt) when available.
__device__ __forceinline__ void copy16_g2l(h16* l, const h16* g) {
#if USE_ASYNC_LDS
    __builtin_amdgcn_global_load_async_to_lds_b128(
        (AS1 v4i*)g, (AS3 v4i*)l, 0, 0);
#else
    *(v8h*)l = *(const v8h*)g;
#endif
}

__device__ __forceinline__ void wait_async() {
#if USE_ASYNC_LDS
#if __has_builtin(__builtin_amdgcn_s_wait_asynccnt)
    __builtin_amdgcn_s_wait_asynccnt(0);
#else
    asm volatile("s_wait_asynccnt 0x0" ::: "memory");
#endif
#endif
}

// Load a 16x32 f16 fragment (A-style == B-style-for-Wt) from a row-major
// buffer with `stride` halves between rows.  Per ISA 16-bit A layout:
// lanes 0-15 (r=lane) hold cols {0..7,16..23}; lanes 16-31 hold {8..15,24..31}.
__device__ __forceinline__ v16h load_frag(const h16* base, int stride) {
    const int lane = threadIdx.x & 31;
    const int r    = lane & 15;
    const int hi   = lane >> 4;
    const h16* p = base + (size_t)r * stride + hi * 8;
    union { v16h v; v8h h[2]; } u;
    u.h[0] = *(const v8h*)(p);
    u.h[1] = *(const v8h*)(p + 16);
    return u.v;
}

// -------------------------------------------------------------------------
// Generic WMMA GEMM:  out = act(scale * (A[M,K] @ W[N,K]^T + bias[N]))
// block tile 128(M) x 128(N), 8 waves; each wave owns 32(M) x 64(N)
// (2x4 accumulators).  Double-buffered LDS, async global->LDS tile fill.
// -------------------------------------------------------------------------
__global__ __launch_bounds__(256) void gemm_wmma(
    const h16* __restrict__ A, const h16* __restrict__ W,
    const float* __restrict__ bias,
    float* __restrict__ outF, h16* __restrict__ outH,
    int M, int N, int K, float scale, int relu)
{
    __shared__ __align__(16) h16 sA[2][128 * 40];
    __shared__ __align__(16) h16 sW[2][128 * 40];

    const int tid  = threadIdx.x;
    const int lane = tid & 31;
    const int w    = tid >> 5;
    const int wm   = w & 3;      // 4 waves along M (32 rows each)
    const int wn   = w >> 2;     // 2 waves along N (64 cols each)
    const int bm   = blockIdx.y * 128;
    const int bn   = blockIdx.x * 128;

    // tile fill: A 128x32 halves (512 x 16B chunks) + W 128x32 (512 chunks)
    const int arow = tid >> 2, ach = tid & 3;   // thread's A/W chunk coords
    const h16* gA0 = A + (size_t)(bm + arow) * K + ach * 8;
    const h16* gA1 = A + (size_t)(bm + arow + 64) * K + ach * 8;
    const h16* gW0 = W + (size_t)(bn + arow) * K + ach * 8;
    const h16* gW1 = W + (size_t)(bn + arow + 64) * K + ach * 8;
    const int lofs0 = arow * 40 + ach * 8;
    const int lofs1 = (arow + 64) * 40 + ach * 8;

    v8f acc[2][4];
#pragma unroll
    for (int i = 0; i < 2; ++i)
#pragma unroll
        for (int j = 0; j < 4; ++j) acc[i][j] = zero8();

    // prologue: fill buffer 0 with k0 = 0
    copy16_g2l(sA[0] + lofs0, gA0);
    copy16_g2l(sA[0] + lofs1, gA1);
    copy16_g2l(sW[0] + lofs0, gW0);
    copy16_g2l(sW[0] + lofs1, gW1);
    wait_async();
    __syncthreads();

    const int nsteps = K >> 5;
    for (int s = 0; s < nsteps; ++s) {
        const int cur = s & 1;
        if (s + 1 < nsteps) {                 // fill the other buffer
            const int k1 = (s + 1) * 32;
            copy16_g2l(sA[cur ^ 1] + lofs0, gA0 + k1);
            copy16_g2l(sA[cur ^ 1] + lofs1, gA1 + k1);
            copy16_g2l(sW[cur ^ 1] + lofs0, gW0 + k1);
            copy16_g2l(sW[cur ^ 1] + lofs1, gW1 + k1);
        }

        // compute from buffer `cur`
        v16h af0 = load_frag(sA[cur] + (wm * 32     ) * 40, 40);
        v16h af1 = load_frag(sA[cur] + (wm * 32 + 16) * 40, 40);
#pragma unroll
        for (int ni = 0; ni < 4; ++ni) {
            v16h bf = load_frag(sW[cur] + (wn * 64 + ni * 16) * 40, 40);
            acc[0][ni] = wmma32(af0, bf, acc[0][ni]);
            acc[1][ni] = wmma32(af1, bf, acc[1][ni]);
        }

        if (s + 1 < nsteps) wait_async();
        __syncthreads();
    }

    // epilogue: C layout: VGPR i -> row i + 8*(lane/16), col = lane%16
    const int r  = lane & 15;
    const int hi = lane >> 4;
#pragma unroll
    for (int mi = 0; mi < 2; ++mi)
#pragma unroll
        for (int ni = 0; ni < 4; ++ni)
#pragma unroll
            for (int i = 0; i < 8; ++i) {
                const int row = bm + wm * 32 + mi * 16 + i + hi * 8;
                const int col = bn + wn * 64 + ni * 16 + r;
                float v = (acc[mi][ni][i] + bias[col]) * scale;
                if (relu) v = v > 0.0f ? v : 0.0f;
                const size_t idx = (size_t)row * N + col;
                if (outF) outF[idx] = v;
                if (outH) outH[idx] = (h16)v;
            }
}

// -------------------------------------------------------------------------
// Flash attention: grid (L/128, H, B), 256 threads = 8 waves x 16 rows.
// Q pre-scaled; clamp +/-50000 then key mask then online softmax.
// -------------------------------------------------------------------------
__global__ __launch_bounds__(256) void attn_wmma(
    const h16* __restrict__ Q, const h16* __restrict__ Kt,
    const h16* __restrict__ V, const int* __restrict__ qmask,
    h16* __restrict__ O)
{
    __shared__ __align__(16) h16 sK [64 * 72];   // K tile  [s][d]
    __shared__ __align__(16) h16 sVT[64 * 72];   // V^T tile [d][s]
    __shared__ __align__(16) h16 sP [128 * 72];  // per-wave P staging
    __shared__ int sMask[64];

    const int tid  = threadIdx.x;
    const int lane = tid & 31;
    const int w    = tid >> 5;
    const int r    = lane & 15;
    const int hi   = lane >> 4;

    const int b  = blockIdx.z;
    const int h  = blockIdx.y;
    const int rb = blockIdx.x;

    const size_t base = ((size_t)b * L_) * D_ + h * DH_;   // [b, l, h*64 + d]

    // this wave's 16 query rows, Q fragments kept in registers
    const h16* qbase = Q + base + (size_t)(rb * 128 + w * 16) * D_;
    v16h qfr[2];
    qfr[0] = load_frag(qbase +  0, D_);   // d 0..31
    qfr[1] = load_frag(qbase + 32, D_);   // d 32..63

    float m_i[8], l_i[8];
    v8f   o_acc[4];
#pragma unroll
    for (int i = 0; i < 8; ++i) { m_i[i] = -3.0e38f; l_i[i] = 0.0f; }
#pragma unroll
    for (int j = 0; j < 4; ++j) o_acc[j] = zero8();

    for (int kb = 0; kb < L_ / 64; ++kb) {
        __syncthreads();
        // K tile via async global->LDS; V tile transposed through registers
#pragma unroll
        for (int it = 0; it < 2; ++it) {
            const int idx = tid + it * 256;        // 512 chunks of 8 halves
            const int s = idx >> 3, ch = idx & 7;
            copy16_g2l(sK + s * 72 + ch * 8,
                       Kt + base + (size_t)(kb * 64 + s) * D_ + ch * 8);
            const h16* gv = V + base + (size_t)(kb * 64 + s) * D_ + ch * 8;
            v8h xv = *(const v8h*)gv;
#pragma unroll
            for (int e = 0; e < 8; ++e) sVT[(ch * 8 + e) * 72 + s] = xv[e];
        }
        if (tid < 64) sMask[tid] = qmask[b * L_ + kb * 64 + tid];
        wait_async();
        __syncthreads();

        // S = Q K^T  (16 rows x 64 keys per wave)
        v8f sacc[4];
#pragma unroll
        for (int j = 0; j < 4; ++j) sacc[j] = zero8();
#pragma unroll
        for (int kst = 0; kst < 2; ++kst) {
#pragma unroll
            for (int j = 0; j < 4; ++j) {
                v16h bf = load_frag(sK + (j * 16) * 72 + kst * 32, 72);
                sacc[j] = wmma32(qfr[kst], bf, sacc[j]);
            }
        }

        // clamp + mask
        float pm[4][8];
#pragma unroll
        for (int j = 0; j < 4; ++j) {
            const bool live = sMask[j * 16 + r] != 0;
#pragma unroll
            for (int i = 0; i < 8; ++i) {
                float v = sacc[j][i];
                v = fminf(fmaxf(v, -CLAMP_), CLAMP_);
                pm[j][i] = live ? v : -1.0e30f;
            }
        }

        // online softmax per row (row = i + 8*hi; reduce across 16 lanes)
#pragma unroll
        for (int i = 0; i < 8; ++i) {
            float mx = pm[0][i];
#pragma unroll
            for (int j = 1; j < 4; ++j) mx = fmaxf(mx, pm[j][i]);
            for (int off = 1; off < 16; off <<= 1)
                mx = fmaxf(mx, __shfl_xor(mx, off, 32));
            const float mnew  = fmaxf(m_i[i], mx);
            const float alpha = __expf(m_i[i] - mnew);
            l_i[i] *= alpha;
#pragma unroll
            for (int j = 0; j < 4; ++j) o_acc[j][i] *= alpha;
            float rs = 0.0f;
#pragma unroll
            for (int j = 0; j < 4; ++j) {
                const float p = __expf(pm[j][i] - mnew);
                pm[j][i] = p;
                rs += p;
            }
            for (int off = 1; off < 16; off <<= 1)
                rs += __shfl_xor(rs, off, 32);
            l_i[i] += rs;
            m_i[i]  = mnew;
        }

        // stage P (f16) in wave-private LDS strip, reload as A-fragments
        h16* pw = sP + (w * 16) * 72;
#pragma unroll
        for (int j = 0; j < 4; ++j)
#pragma unroll
            for (int i = 0; i < 8; ++i)
                pw[(i + hi * 8) * 72 + j * 16 + r] = (h16)pm[j][i];

        v16h pf0 = load_frag(pw +  0, 72);
        v16h pf1 = load_frag(pw + 32, 72);
#pragma unroll
        for (int jd = 0; jd < 4; ++jd) {
            v16h v0 = load_frag(sVT + (jd * 16) * 72 +  0, 72);
            v16h v1 = load_frag(sVT + (jd * 16) * 72 + 32, 72);
            o_acc[jd] = wmma32(pf0, v0, o_acc[jd]);
            o_acc[jd] = wmma32(pf1, v1, o_acc[jd]);
        }
    }

    // normalize and store (B, L, H*64) f16
#pragma unroll
    for (int i = 0; i < 8; ++i) {
        const float inv = 1.0f / l_i[i];
        const int row = rb * 128 + w * 16 + i + hi * 8;
#pragma unroll
        for (int jd = 0; jd < 4; ++jd) {
            const int col = jd * 16 + r;
            O[base + (size_t)row * D_ + col] = (h16)(o_acc[jd][i] * inv);
        }
    }
}

// -------------------------------------------------------------------------
// residual add + LayerNorm over 512 cols, one row per 256-thread block
// -------------------------------------------------------------------------
__global__ __launch_bounds__(256) void ln_kernel(
    const float* __restrict__ A, const float* __restrict__ R,
    const float* __restrict__ g, const float* __restrict__ be,
    float* __restrict__ outF, h16* __restrict__ outH)
{
    __shared__ float rs[256], rs2[256];
    const int row = blockIdx.x;
    const int tid = threadIdx.x;
    const size_t baseR = (size_t)row * D_;

    const float x0 = A[baseR + tid]       + R[baseR + tid];
    const float x1 = A[baseR + 256 + tid] + R[baseR + 256 + tid];
    rs [tid] = x0 + x1;
    rs2[tid] = x0 * x0 + x1 * x1;
    __syncthreads();
    for (int o = 128; o > 0; o >>= 1) {
        if (tid < o) { rs[tid] += rs[tid + o]; rs2[tid] += rs2[tid + o]; }
        __syncthreads();
    }
    const float mu  = rs[0] * (1.0f / D_);
    const float var = rs2[0] * (1.0f / D_) - mu * mu;
    const float inv = rsqrtf(var + 1e-5f);

    const float y0 = (x0 - mu) * inv * g[tid]       + be[tid];
    const float y1 = (x1 - mu) * inv * g[256 + tid] + be[256 + tid];
    outF[baseR + tid]       = y0;
    outF[baseR + 256 + tid] = y1;
    if (outH) {
        outH[baseR + tid]       = (h16)y0;
        outH[baseR + 256 + tid] = (h16)y1;
    }
}

// -------------------------------------------------------------------------
// conversion kernels
// -------------------------------------------------------------------------
__global__ void cvt_f32_f16(const float* __restrict__ in, h16* __restrict__ out, int n) {
    const int i = blockIdx.x * 256 + threadIdx.x;
    if (i < n) out[i] = (h16)in[i];
}

__global__ void addcvt(const float* __restrict__ q, const float* __restrict__ qp,
                       h16* __restrict__ qp16, h16* __restrict__ q16, int n) {
    const int i = blockIdx.x * 256 + threadIdx.x;
    if (i < n) {
        const float a = q[i];
        qp16[i] = (h16)(a + qp[i]);
        q16[i]  = (h16)a;
    }
}

// -------------------------------------------------------------------------
// launch
// -------------------------------------------------------------------------
extern "C" void kernel_launch(void* const* d_in, const int* in_sizes, int n_in,
                              void* d_out, int out_size, void* d_ws, size_t ws_size,
                              hipStream_t stream) {
    const float* query = (const float*)d_in[0];
    const int*   qmask = (const int*)  d_in[1];
    const float* qpos  = (const float*)d_in[2];
    const float* Wq    = (const float*)d_in[3];
    const float* bq    = (const float*)d_in[4];
    const float* Wk    = (const float*)d_in[5];
    const float* bk    = (const float*)d_in[6];
    const float* Wv    = (const float*)d_in[7];
    const float* bv    = (const float*)d_in[8];
    const float* Wo    = (const float*)d_in[9];
    const float* bo    = (const float*)d_in[10];
    const float* W1    = (const float*)d_in[11];
    const float* b1    = (const float*)d_in[12];
    const float* W2    = (const float*)d_in[13];
    const float* b2    = (const float*)d_in[14];
    const float* g1    = (const float*)d_in[15];
    const float* be1   = (const float*)d_in[16];
    const float* g2    = (const float*)d_in[17];
    const float* be2   = (const float*)d_in[18];

    char* ws = (char*)d_ws;
    size_t off = 0;
    auto alloc = [&](size_t bytes) {
        size_t o = off;
        off += (bytes + 255) & ~(size_t)255;
        return o;
    };

    h16* wq16 = (h16*)(ws + alloc((size_t)D_ * D_ * 2));
    h16* wk16 = (h16*)(ws + alloc((size_t)D_ * D_ * 2));
    h16* wv16 = (h16*)(ws + alloc((size_t)D_ * D_ * 2));
    h16* wo16 = (h16*)(ws + alloc((size_t)D_ * D_ * 2));
    h16* w116 = (h16*)(ws + alloc((size_t)HID_ * D_ * 2));
    h16* w216 = (h16*)(ws + alloc((size_t)D_ * HID_ * 2));
    h16* qp16 = (h16*)(ws + alloc((size_t)NT_ * D_ * 2));
    h16* qr16 = (h16*)(ws + alloc((size_t)NT_ * D_ * 2));
    h16* q16  = (h16*)(ws + alloc((size_t)NT_ * D_ * 2));
    h16* k16  = (h16*)(ws + alloc((size_t)NT_ * D_ * 2));
    h16* v16  = (h16*)(ws + alloc((size_t)NT_ * D_ * 2));
    h16* at16 = (h16*)(ws + alloc((size_t)NT_ * D_ * 2));
    h16* x16  = (h16*)(ws + alloc((size_t)NT_ * D_ * 2));
    h16* h16b = (h16*)(ws + alloc((size_t)NT_ * HID_ * 2));
    float* tmp32 = (float*)(ws + alloc((size_t)NT_ * D_ * 4));
    float* x32   = (float*)(ws + alloc((size_t)NT_ * D_ * 4));

    // weight conversion
    cvt_f32_f16<<<(D_ * D_ + 255) / 256, 256, 0, stream>>>(Wq, wq16, D_ * D_);
    cvt_f32_f16<<<(D_ * D_ + 255) / 256, 256, 0, stream>>>(Wk, wk16, D_ * D_);
    cvt_f32_f16<<<(D_ * D_ + 255) / 256, 256, 0, stream>>>(Wv, wv16, D_ * D_);
    cvt_f32_f16<<<(D_ * D_ + 255) / 256, 256, 0, stream>>>(Wo, wo16, D_ * D_);
    cvt_f32_f16<<<(HID_ * D_ + 255) / 256, 256, 0, stream>>>(W1, w116, HID_ * D_);
    cvt_f32_f16<<<(D_ * HID_ + 255) / 256, 256, 0, stream>>>(W2, w216, D_ * HID_);

    // qp = query + pos (f16), query (f16)
    addcvt<<<(NT_ * D_ + 255) / 256, 256, 0, stream>>>(query, qpos, qp16, qr16, NT_ * D_);

    const dim3 blk(256);
    // projections (Q carries the 1/sqrt(64) scale after bias, matching reference)
    gemm_wmma<<<dim3(D_ / 128, NT_ / 128), blk, 0, stream>>>(
        qp16, wq16, bq, nullptr, q16, NT_, D_, D_, 0.125f, 0);
    gemm_wmma<<<dim3(D_ / 128, NT_ / 128), blk, 0, stream>>>(
        qp16, wk16, bk, nullptr, k16, NT_, D_, D_, 1.0f, 0);
    gemm_wmma<<<dim3(D_ / 128, NT_ / 128), blk, 0, stream>>>(
        qr16, wv16, bv, nullptr, v16, NT_, D_, D_, 1.0f, 0);

    // attention
    attn_wmma<<<dim3(L_ / 128, H_, B_), blk, 0, stream>>>(q16, k16, v16, qmask, at16);

    // output projection -> f32, then residual + LN1 (emits f32 + f16)
    gemm_wmma<<<dim3(D_ / 128, NT_ / 128), blk, 0, stream>>>(
        at16, wo16, bo, tmp32, nullptr, NT_, D_, D_, 1.0f, 0);
    ln_kernel<<<NT_, blk, 0, stream>>>(tmp32, query, g1, be1, x32, x16);

    // FFN
    gemm_wmma<<<dim3(HID_ / 128, NT_ / 128), blk, 0, stream>>>(
        x16, w116, b1, nullptr, h16b, NT_, HID_, D_, 1.0f, 1);
    gemm_wmma<<<dim3(D_ / 128, NT_ / 128), blk, 0, stream>>>(
        h16b, w216, b2, tmp32, nullptr, NT_, D_, HID_, 1.0f, 0);

    // residual + LN2 -> final f32 output
    ln_kernel<<<NT_, blk, 0, stream>>>(tmp32, x32, g2, be2, (float*)d_out, nullptr);
}
